// Split_84851373900209
// MI455X (gfx1250) — matrix operand order repaired and checked by reading, not compiled, not run on hardware
//
#include <hip/hip_runtime.h>
#include <hip/hip_bf16.h>

typedef __attribute__((ext_vector_type(16))) _Float16 v16h;
typedef __attribute__((ext_vector_type(8)))  _Float16 v8h;
typedef __attribute__((ext_vector_type(4)))  _Float16 v4h;
typedef __attribute__((ext_vector_type(8)))  float    v8f;

#define Bn 4096
#define Nn 32
#define In 32
#define Hn 128
#define Ln 2
#define Cn (Hn + In)   // 160

// ---------------- fragment helpers (CDNA5 ISA 7.12.2, wave32) ----------------

// A-matrix 16x32 f16 from row-major f16 LDS (stride in halfs).
__device__ __forceinline__ v16h ldsA(const _Float16* base, int stride, int mt, int kt, int lane) {
    int row = mt * 16 + (lane & 15);
    int kc  = kt * 32 + ((lane & 16) ? 16 : 0);
    const _Float16* p = base + row * stride + kc;
    v16h a;
#pragma unroll
    for (int i = 0; i < 8; ++i) a[i] = p[i];
#pragma unroll
    for (int i = 0; i < 8; ++i) a[8 + i] = p[8 + i];
    return a;
}

// B-matrix 32x16 f16 from COLUMN-major f16 LDS: layout [col][k].
__device__ __forceinline__ v16h ldsBT(const _Float16* base, int kdim, int nt, int kt, int lane) {
    int col = nt * 16 + (lane & 15);
    int kb  = kt * 32 + ((lane & 16) ? 16 : 0);
    const _Float16* p = base + col * kdim + kb;
    v16h b;
#pragma unroll
    for (int i = 0; i < 16; ++i) b[i] = p[i];
    return b;
}

// W fragment from pre-swizzled f16 workspace: frag*512 + lane*16, contiguous 32B.
__device__ __forceinline__ v16h wsB(const _Float16* ws, int frag, int lane) {
    const _Float16* p = ws + ((size_t)frag << 9) + (lane << 4);
    v16h b;
#pragma unroll
    for (int i = 0; i < 16; ++i) b[i] = p[i];
    return b;
}

__device__ __forceinline__ v8f wmma16(v16h a, v16h b, v8f c) {
    return __builtin_amdgcn_wmma_f32_16x16x32_f16(false, a, false, b, (short)0, c, false, false);
}

// ---------------- pre-pass: swizzle W1/W2 (f32) into f16 B-fragment layout ----------------
// frag index = ((l*2 + mat)*8 + nt)*5 + k   (160 fragments x 512 halfs)
__global__ void __launch_bounds__(256)
swizzle_weights(const float* __restrict__ W1, const float* __restrict__ W2,
                _Float16* __restrict__ wsW) {
    int idx = blockIdx.x * 256 + threadIdx.x;          // 0 .. 160*512-1
    int i    = idx & 15;
    int lane = (idx >> 4) & 31;
    int f    = idx >> 9;
    int k    = f % 5;
    int nt   = (f / 5) & 7;
    int mat  = (f / 40) & 1;
    int l    = f / 80;
    int col  = nt * 16 + (lane & 15);
    int krow = k * 32 + ((lane & 16) ? 16 : 0) + i;
    const float* W = mat ? W2 : W1;
    wsW[idx] = (_Float16)W[((size_t)l * Cn + krow) * Hn + col];
}

// ---------------- main fused kernel: one wave32 per batch ----------------
__global__ void __launch_bounds__(32)
fused_gnn_forward(const float* __restrict__ e, const float* __restrict__ x,
                  const float* __restrict__ mask, const float* __restrict__ hid0,
                  const _Float16* __restrict__ wsW,
                  const float* __restrict__ bias, const float* __restrict__ wb,
                  const float* __restrict__ bb, const int* __restrict__ scale,
                  float* __restrict__ out_scores, float* __restrict__ out_probs,
                  float* __restrict__ out_inpn, float* __restrict__ out_phi) {
    extern __shared__ char smem[];
    _Float16* sH     = (_Float16*)(smem);            // [32][160] row-major h (A operand)
    _Float16* sTT    = (_Float16*)(smem + 10240);    // [128][32] col-major T = h@W2 (B operand)
    float*    sMean  = (float*)   (smem + 10240);    // [32][32]  (aliases sTT, used before layers)
    float*    sEx2   = (float*)   (smem + 14336);    // [32][32]  (aliases sTT, used before layers)
    _Float16* sXT    = (_Float16*)(smem + 18432);    // [32][32]  col-major x
    _Float16* sPhi   = (_Float16*)(smem + 20480);    // [32][32]  row-major raw Phi (f16, exact 0/1)
    float*    sNvInv = (float*)   (smem + 22528);    // [32]

    const int b = blockIdx.x;
    const int t = threadIdx.x;                       // lane; one wave per batch
    const size_t bn = (size_t)b * Nn;

    const int sc = scale[0];
    float ev = (sc == 0) ? 0.0f : e[bn + t];
    float mv = mask[bn + t];

    // ---- Phi row t: raw (exact 0/1) to output + f16 LDS, accumulate Nv ----
    float nv = 0.0f;
#pragma unroll
    for (int j = 0; j < 32; ++j) {
        float ej = __shfl(ev, j, 32);
        float mj = __shfl(mv, j, 32);
        float p  = (ev == ej) ? (mv * mj) : 0.0f;
        out_phi[(bn + t) * Nn + j] = p;
        sPhi[t * 32 + j] = (_Float16)p;
        nv += p;
    }
    if (nv == 0.0f) nv = 1.0f;
    sNvInv[t] = __builtin_amdgcn_rcpf(nv);

    // ---- stage x (col-major f16) and hidden0 (row-major f16), vectorized ----
    {
        const float4* x4 = (const float4*)(x + bn * In);          // 256 chunks
        for (int q = t; q < 256; q += 32) {
            float4 v = x4[q];
            int r = q >> 3, c = (q & 7) * 4;                      // elem idx = q*4
            sXT[(c + 0) * 32 + r] = (_Float16)v.x;
            sXT[(c + 1) * 32 + r] = (_Float16)v.y;
            sXT[(c + 2) * 32 + r] = (_Float16)v.z;
            sXT[(c + 3) * 32 + r] = (_Float16)v.w;
        }
        const float4* h4 = (const float4*)(hid0 + bn * Hn);       // 1024 floats, 256 chunks
        for (int q = t; q < 256; q += 32) {
            float4 v = h4[q];
            int r = q >> 5, c = (q & 31) * 4;
            v4h hv = {(_Float16)v.x, (_Float16)v.y, (_Float16)v.z, (_Float16)v.w};
            *(v4h*)(sH + r * Cn + c) = hv;
        }
    }

    // ---- A fragments of Phi' = diag(1/Nv)*Phi: load raw, scale own row (v_pk_mul_f16) ----
    v16h aPhi0 = ldsA(sPhi, 32, 0, 0, t);
    v16h aPhi1 = ldsA(sPhi, 32, 1, 0, t);
    {
        _Float16 s0 = (_Float16)sNvInv[(t & 15)];
        _Float16 s1 = (_Float16)sNvInv[16 + (t & 15)];
#pragma unroll
        for (int i = 0; i < 16; ++i) { aPhi0[i] *= s0; aPhi1[i] *= s1; }
    }
    const int colL = (t & 15);
    const int hi8  = (t & 16) ? 8 : 0;

    // ---- means / Ex2 via WMMA (already /Nv): Phi' @ x, Phi' @ x^2 ----
#pragma unroll
    for (int nt = 0; nt < 2; ++nt) {
        v16h bx = ldsBT(sXT, 32, nt, 0, t);
        v16h bx2;
#pragma unroll
        for (int i = 0; i < 16; ++i) bx2[i] = bx[i] * bx[i];
        v8f z = {};
        v8f dm0 = wmma16(aPhi0, bx,  z);
        v8f dm1 = wmma16(aPhi1, bx,  z);
        v8f d20 = wmma16(aPhi0, bx2, z);
        v8f d21 = wmma16(aPhi1, bx2, z);
        int col = nt * 16 + colL;
#pragma unroll
        for (int v = 0; v < 8; ++v) {
            sMean[(hi8 + v) * 32 + col]      = dm0[v];
            sMean[(16 + hi8 + v) * 32 + col] = dm1[v];
            sEx2 [(hi8 + v) * 32 + col]      = d20[v];
            sEx2 [(16 + hi8 + v) * 32 + col] = d21[v];
        }
    }

    // ---- input_n row t (vectorized float4) ----
    {
        const float4* x4 = (const float4*)(x + (bn + t) * In);
        float4*       o4 = (float4*)(out_inpn + (bn + t) * In);
        const float4* m4 = (const float4*)(sMean + t * 32);
        const float4* e4 = (const float4*)(sEx2 + t * 32);
#pragma unroll
        for (int q = 0; q < 8; ++q) {
            float4 xv = x4[q], mn = m4[q], ex = e4[q];
            float r0[4] = {xv.x, xv.y, xv.z, xv.w};
            float r1[4] = {mn.x, mn.y, mn.z, mn.w};
            float r2[4] = {ex.x, ex.y, ex.z, ex.w};
            float4 o;
            v4h hv;
#pragma unroll
            for (int i = 0; i < 4; ++i) {
                float var = r2[i] - r1[i] * r1[i];
                var = fmaxf(var, 0.0f);
                if (var == 0.0f) var = 1.0f;
                float inp = ((r0[i] - r1[i]) * (rsqrtf(var) * (1.0f / 3.0f)) + 0.5f) * mv;
                ((float*)&o)[i] = inp;
                hv[i] = (_Float16)inp;
            }
            o4[q] = o;
            *(v4h*)(sH + t * Cn + Hn + q * 4) = hv;
        }
    }

    // ---- L layers: hidden = sigmoid(h@W1 + Phi' @ (h@W2) + b) ----
    for (int l = 0; l < Ln; ++l) {
        // pass 1: T = h @ W2[l], col-major (B-ready) in sTT.
        // Hold A frags for BOTH row-tiles; each W fragment feeds two WMMAs.
        {
            v16h aH0[5], aH1[5];
#pragma unroll
            for (int k = 0; k < 5; ++k) {
                aH0[k] = ldsA(sH, Cn, 0, k, t);
                aH1[k] = ldsA(sH, Cn, 1, k, t);
            }
            for (int nt = 0; nt < 8; ++nt) {
                v8f a0 = {}, a1 = {};
#pragma unroll
                for (int k = 0; k < 5; ++k) {
                    v16h b2 = wsB(wsW, ((l * 2 + 1) * 8 + nt) * 5 + k, t);
                    a0 = wmma16(aH0[k], b2, a0);
                    a1 = wmma16(aH1[k], b2, a1);
                }
                int col = nt * 16 + colL;
                v8h h0, h1;
#pragma unroll
                for (int v = 0; v < 8; ++v) { h0[v] = (_Float16)a0[v]; h1[v] = (_Float16)a1[v]; }
                *(v8h*)(sTT + col * Nn + hi8)      = h0;   // rows hi8..hi8+7
                *(v8h*)(sTT + col * Nn + 16 + hi8) = h1;   // rows 16+hi8..
            }
        }
        // pass 2: acc = h@W1[l] (K=160) + Phi'@T (K=32); bias+sigmoid; write into sH.
        {
            v16h aH0[5], aH1[5];
#pragma unroll
            for (int k = 0; k < 5; ++k) {
                aH0[k] = ldsA(sH, Cn, 0, k, t);
                aH1[k] = ldsA(sH, Cn, 1, k, t);
            }
            for (int nt = 0; nt < 8; ++nt) {
                v8f a0 = {}, a1 = {};
#pragma unroll
                for (int k = 0; k < 5; ++k) {
                    v16h b1 = wsB(wsW, ((l * 2 + 0) * 8 + nt) * 5 + k, t);
                    a0 = wmma16(aH0[k], b1, a0);
                    a1 = wmma16(aH1[k], b1, a1);
                }
                v16h bt = ldsBT(sTT, Nn, nt, 0, t);
                a0 = wmma16(aPhi0, bt, a0);
                a1 = wmma16(aPhi1, bt, a1);
                int col = nt * 16 + colL;
                float bc = bias[l * Hn + col];
#pragma unroll
                for (int v = 0; v < 8; ++v) {
                    float v0 = __builtin_amdgcn_rcpf(1.0f + __expf(-(a0[v] + bc)));
                    float v1 = __builtin_amdgcn_rcpf(1.0f + __expf(-(a1[v] + bc)));
                    sH[(hi8 + v) * Cn + col]      = (_Float16)v0;
                    sH[(16 + hi8 + v) * Cn + col] = (_Float16)v1;
                }
            }
        }
    }

    // ---- scores / probs (row t), vectorized v8h reads ----
    float s = bb[0];
    const v8h* hrow = (const v8h*)(sH + t * Cn);   // row byte offset t*320, 16B aligned
#pragma unroll
    for (int q = 0; q < 16; ++q) {                 // 16 x 8 = 128 hidden cols
        v8h hv = hrow[q];
#pragma unroll
        for (int i = 0; i < 8; ++i) s += (float)hv[i] * wb[q * 8 + i];
    }
    out_scores[bn + t] = s;
    out_probs [bn + t] = __builtin_amdgcn_rcpf(1.0f + __expf(-s)) * mv;
}

extern "C" void kernel_launch(void* const* d_in, const int* in_sizes, int n_in,
                              void* d_out, int out_size, void* d_ws, size_t ws_size,
                              hipStream_t stream) {
    const float* e    = (const float*)d_in[0];
    const float* x    = (const float*)d_in[1];
    const float* mask = (const float*)d_in[2];
    const float* hid0 = (const float*)d_in[3];
    const float* W1   = (const float*)d_in[4];
    const float* W2   = (const float*)d_in[5];
    const float* bias = (const float*)d_in[6];
    const float* wb   = (const float*)d_in[7];
    const float* bb   = (const float*)d_in[8];
    const int*   scl  = (const int*)d_in[9];

    float* out = (float*)d_out;
    const size_t BN = (size_t)Bn * Nn;
    float* out_scores = out;
    float* out_probs  = out + BN;
    float* out_inpn   = out + 2 * BN;
    float* out_phi    = out + 2 * BN + BN * In;

    _Float16* wsW = (_Float16*)d_ws;                 // 160 frags * 512 halfs = 160 KB

    // pre-pass: W1/W2 f32 -> f16 fragment-swizzled workspace (runs every call)
    swizzle_weights<<<(Ln * 2 * 8 * 5 * 512) / 256, 256, 0, stream>>>(W1, W2, wsW);

    int nblocks = in_sizes[0] / Nn;                  // = B
    size_t smem = 22656;   // sH 10240 | sTT(sMean+sEx2) 8192 | sXT 2048 | sPhi 2048 | sNvInv 128

    fused_gnn_forward<<<nblocks, 32, smem, stream>>>(
        e, x, mask, hid0, wsW, bias, wb, bb, scl,
        out_scores, out_probs, out_inpn, out_phi);
}
